// SceneGraphEquiModel_67671504715834
// MI455X (gfx1250) — matrix operand order repaired and checked by reading, not compiled, not run on hardware
//
#include <hip/hip_runtime.h>
#include <hip/hip_bf16.h>
#include <math.h>

// ---------------- CDNA5 WMMA / TDM types ----------------
typedef __attribute__((ext_vector_type(16))) __bf16 v16bf;
typedef __attribute__((ext_vector_type(8)))  float  v8f;
typedef __attribute__((ext_vector_type(4)))  unsigned v4u;
typedef __attribute__((ext_vector_type(8)))  int      v8i;
typedef __attribute__((ext_vector_type(4)))  int      v4i;

#if defined(__has_builtin)
#if __has_builtin(__builtin_amdgcn_tensor_load_to_lds) && __has_builtin(__builtin_amdgcn_s_wait_tensorcnt)
#define USE_TDM 1
#endif
#endif
#ifndef USE_TDM
#define USE_TDM 0
#endif

// ---------------- workspace layout (float offsets) ----------------
static const long O_CONV1 = 0;           // 25,690,112  (4*128*224*224)
static const long O_POOL1 = 25690112;    //  6,422,528  (4*112*112*128 NHWC)
static const long O_CONV3 = 32112640;    //    802,816  (4*64*56*56)
static const long O_H1    = 32915456;    //  1,048,576  (4096*256)
static const long O_H2    = 33964032;    //  1,048,576
static const long O_G1H   = 35012608;    //  4,194,304  (4096*1024)
static const long O_AGG1  = 39206912;    //  4,194,304
static const long O_G2H   = 43401216;    //  1,048,576
static const long O_HFIN  = 44449792;    //  1,048,576
static const long O_ES1   = 45498368;    //     16,384  (4096*4)
static const long O_ED1   = 45514752;
static const long O_MAX1  = 45531136;
static const long O_DEN1  = 45547520;
static const long O_EATT1 = 45563904;    //    278,528  (69632*4)
static const long O_ES2   = 45842432;    //      4,096
static const long O_ED2   = 45846528;
static const long O_MAX2  = 45850624;
static const long O_DEN2  = 45854720;
static const long O_EATT2 = 45858816;    //     69,632
static const long O_GN    = 45928448;    //        256  group-norm stats
static const long O_WBF   = 45928704;    //  884,352 floats = 1,768,704 bf16 weights
// total ~46.8M floats = ~187 MB

// bf16 weight pool element offsets
static const long WB_C2 = 0;        // 294,912 (permuted conv2)
static const long WB_C3 = 294912;   //  16,384
static const long WB_M1 = 311296;   // 802,816
static const long WB_M2 = 1114112;  //  65,536
static const long WB_G1 = 1179648;  // 262,144
static const long WB_G2 = 1441792;  // 262,144
static const long WB_OB = 1703936;  //  38,656
static const long WB_RL = 1742592;  //  26,112

static const int NB   = 4;
static const int NBOX = 4096;
static const int NEDG = 65536;
static const int ETOT = NEDG + NBOX;     // edges + self loops

// ---------------- small utility kernels ----------------
__global__ void fill_kernel(float* p, long n, float v) {
  long i = (long)blockIdx.x * blockDim.x + threadIdx.x;
  if (i < n) p[i] = v;
}

__global__ void bias_add_kernel(float* x, const float* b, long n, int F) {
  long i = (long)blockIdx.x * blockDim.x + threadIdx.x;
  if (i < n) x[i] += b[i % F];
}

__global__ void f32_to_bf16_kernel(const float* __restrict__ in, __bf16* __restrict__ out, long n) {
  long i = (long)blockIdx.x * blockDim.x + threadIdx.x;
  if (i < n) out[i] = (__bf16)in[i];
}

// conv2 weight permute+convert: out[n][tap*128+ci] = (bf16)w[n][ci*9+tap]   (K = 1152)
__global__ void permute_wc2_kernel(const float* __restrict__ w, __bf16* __restrict__ out) {
  int i = blockIdx.x * blockDim.x + threadIdx.x;
  if (i >= 256 * 1152) return;
  int n = i / 1152, r = i - n * 1152;
  int tap = r >> 7, ci = r & 127;
  out[n * 1152 + tap * 128 + ci] = (__bf16)w[n * 1152 + ci * 9 + tap];
}

// ---------------- conv1: direct 3x3, Cin=3 ----------------
__global__ void conv1_kernel(const float* __restrict__ img, const float* __restrict__ w,
                             float* __restrict__ out, int B, int Cout, int H, int W) {
  long idx = (long)blockIdx.x * blockDim.x + threadIdx.x;
  long tot = (long)B * Cout * H * W;
  if (idx >= tot) return;
  int ow = idx % W; long t = idx / W;
  int oh = t % H;   t /= H;
  int co = t % Cout; int b = t / Cout;
  float acc = 0.f;
  for (int ci = 0; ci < 3; ++ci)
    for (int kh = 0; kh < 3; ++kh) {
      int ih = oh + kh - 1;
      if ((unsigned)ih >= (unsigned)H) continue;
      for (int kw = 0; kw < 3; ++kw) {
        int iw = ow + kw - 1;
        if ((unsigned)iw >= (unsigned)W) continue;
        acc += img[(((long)b * 3 + ci) * H + ih) * W + iw] *
               w[((co * 3 + ci) * 3 + kh) * 3 + kw];
      }
    }
  out[idx] = acc;
}

// ---------------- Gaussian 5x5 depthwise, stride 2, pad 2; NCHW in -> NHWC out ----------------
__global__ void gauss_pool_nhwc_kernel(const float* __restrict__ x, float* __restrict__ y,
                                       int B, int C, int H, int W) {
  int Ho = H / 2, Wo = W / 2;
  long idx = (long)blockIdx.x * blockDim.x + threadIdx.x;
  long tot = (long)B * C * Ho * Wo;
  if (idx >= tot) return;
  int ow = idx % Wo; long t = idx / Wo;
  int oh = t % Ho;   t /= Ho;
  int c  = t % C;    int b = t / C;
  const float s2 = 2.f * 0.66f * 0.66f;
  float k1[5]; float S = 0.f;
  #pragma unroll
  for (int i = 0; i < 5; ++i) { float r = (float)i - 2.f; k1[i] = __expf(-r * r / s2); S += k1[i]; }
  float inv = 1.f / (S * S);
  const float* xp = x + ((long)b * C + c) * H * W;
  float acc = 0.f;
  #pragma unroll
  for (int ky = 0; ky < 5; ++ky) {
    int iy = oh * 2 + ky - 2;
    if ((unsigned)iy >= (unsigned)H) continue;
    #pragma unroll
    for (int kx = 0; kx < 5; ++kx) {
      int ix = ow * 2 + kx - 2;
      if ((unsigned)ix >= (unsigned)W) continue;
      acc += xp[(long)iy * W + ix] * k1[ky] * k1[kx];
    }
  }
  y[(((long)b * Ho + oh) * Wo + ow) * C + c] = acc * inv;   // NHWC
}

// ---------------- group-norm (inner_bn): stats + apply (NCHW) ----------------
__global__ void gn_accum_kernel(const float* __restrict__ x, float* __restrict__ stats,
                                int B, int C, int HW, int fs) {
  int g = blockIdx.y;
  long cnt = (long)B * fs * HW;
  float s = 0.f, ss = 0.f;
  for (long j = (long)blockIdx.x * blockDim.x + threadIdx.x; j < cnt;
       j += (long)gridDim.x * blockDim.x) {
    long b = j / ((long)fs * HW);
    long r = j - b * (long)fs * HW;
    int ci = (int)(r / HW);
    int hw = (int)(r - (long)ci * HW);
    float v = x[((b * C) + (long)g * fs + ci) * HW + hw];
    s += v; ss += v * v;
  }
  __shared__ float sh[512];
  sh[threadIdx.x] = s; sh[threadIdx.x + 256] = ss;
  __syncthreads();
  for (int off = 128; off > 0; off >>= 1) {
    if ((int)threadIdx.x < off) {
      sh[threadIdx.x] += sh[threadIdx.x + off];
      sh[threadIdx.x + 256] += sh[threadIdx.x + 256 + off];
    }
    __syncthreads();
  }
  if (threadIdx.x == 0) {
    atomicAdd(&stats[g * 2 + 0], sh[0]);
    atomicAdd(&stats[g * 2 + 1], sh[256]);
  }
}

__global__ void gn_apply_kernel(float* __restrict__ x, const float* __restrict__ stats,
                                const float* __restrict__ gamma, const float* __restrict__ beta,
                                int B, int C, int HW, int fs) {
  long idx = (long)blockIdx.x * blockDim.x + threadIdx.x;
  long tot = (long)B * C * HW;
  if (idx >= tot) return;
  long t = idx / HW; int c = (int)(t % C);
  int g = c / fs;
  float cnt = (float)((long)B * fs * HW);
  float mean = stats[g * 2] / cnt;
  float var = stats[g * 2 + 1] / cnt - mean * mean;
  float v = (x[idx] - mean) * rsqrtf(var + 1e-5f) * gamma[g] + beta[g];
  x[idx] = fmaxf(v, 0.f);   // all three BNs are followed by ReLU
}

// ---------------- ROI align (out=7, ratio=2, scale=0.25), NCHW features ----------------
__global__ void roi_align_kernel(const float* __restrict__ feat, const float* __restrict__ boxes,
                                 const int* __restrict__ bidx, float* __restrict__ out,
                                 int C, int H, int W) {
  long idx = (long)blockIdx.x * blockDim.x + threadIdx.x;
  long tot = (long)NBOX * C * 49;
  if (idx >= tot) return;
  int r = (int)(idx % (C * 49));
  int box = (int)(idx / (C * 49));
  int c = r / 49; int oy = (r % 49) / 7; int ox = r % 7;
  float x1 = boxes[box * 4 + 0] * 0.25f - 0.5f;
  float y1 = boxes[box * 4 + 1] * 0.25f - 0.5f;
  float x2 = boxes[box * 4 + 2] * 0.25f - 0.5f;
  float y2 = boxes[box * 4 + 3] * 0.25f - 0.5f;
  float bw = (x2 - x1) / 7.f, bh = (y2 - y1) / 7.f;
  int b = bidx[box];
  const float* fp = feat + ((long)b * C + c) * H * W;
  float acc = 0.f;
  #pragma unroll
  for (int sy = 0; sy < 2; ++sy) {
    float gy = ((float)(oy * 2 + sy) + 0.5f) * 0.5f;
    float ys = y1 + bh * gy;
    bool vy = (ys >= -1.f) && (ys <= (float)H);
    float y = fminf(fmaxf(ys, 0.f), (float)(H - 1));
    float y0f = floorf(y); int y0 = (int)y0f; int yp = min(y0 + 1, H - 1);
    float ly = y - y0f, hy = 1.f - ly;
    #pragma unroll
    for (int sx = 0; sx < 2; ++sx) {
      float gx = ((float)(ox * 2 + sx) + 0.5f) * 0.5f;
      float xs = x1 + bw * gx;
      bool vx = (xs >= -1.f) && (xs <= (float)W);
      float xq = fminf(fmaxf(xs, 0.f), (float)(W - 1));
      float x0f = floorf(xq); int x0 = (int)x0f; int xp = min(x0 + 1, W - 1);
      float lx = xq - x0f, hx = 1.f - lx;
      float v = fp[(long)y0 * W + x0] * hy * hx + fp[(long)y0 * W + xp] * hy * lx +
                fp[(long)yp * W + x0] * ly * hx + fp[(long)yp * W + xp] * ly * lx;
      acc += (vy && vx) ? v : 0.f;
    }
  }
  out[(long)box * (C * 49) + c * 49 + oy * 7 + ox] = acc * 0.25f;
}

// ---------------- generic WMMA GEMM: C = act(A @ W^T + bias), W pre-converted bf16 ----------------
// Invariants (all call sites): K % 32 == 0, M % 128 == 0. Only N may be ragged.
// AMODE: 0 row-major | 1 NHWC im2col 3x3 pad1 (Cin=128, K=tap*128+ci) | 3 edge-concat gather
// CMODE: 0 row-major (ldc) | 1 NCHW
struct GemmP {
  const float* A; const __bf16* B; const float* bias; float* C;
  int M, N, K, lda, ldc;
  int cmode, relu;
  int Hin, Win, HWin;
  const int* e0; const int* e1;
};

__device__ __forceinline__ void cvt8(v16bf& a, int base, float4 qa, float4 qb) {
  a[base+0]=(__bf16)qa.x; a[base+1]=(__bf16)qa.y; a[base+2]=(__bf16)qa.z; a[base+3]=(__bf16)qa.w;
  a[base+4]=(__bf16)qb.x; a[base+5]=(__bf16)qb.y; a[base+6]=(__bf16)qb.z; a[base+7]=(__bf16)qb.w;
}

template <int AMODE>
__global__ __launch_bounds__(256) void gemm_wmma_kernel(GemmP p) {
  // B tile in LDS, shared by all 8 waves: [col 0..63][k 0..31] bf16 = 4KB
  __shared__ __bf16 Bs[64][32];

  const int lane = threadIdx.x & 31;
  const int wave = threadIdx.x >> 5;
  const int half = lane >> 4;       // lane group (0: lanes 0-15, 1: 16-31)
  const int l15  = lane & 15;
  const int mbase = blockIdx.y * 128 + wave * 16;
  const int nbase = blockIdx.x * 64;

  v8f acc[4];
  #pragma unroll
  for (int s = 0; s < 4; ++s)
    #pragma unroll
    for (int j = 0; j < 8; ++j) acc[s][j] = 0.f;

  // A-layout (contiguous, matches B): lanes<16 hold K {0..15}; lanes>=16 hold {16..31}
  const int ka0 = half ? 16 : 0;
  const int ka1 = half ? 24 : 8;
  const int kb  = half ? 16 : 0;   // B 32x16: lanes<16 K 0..15, lanes>=16 K 16..31
  const int am  = mbase + l15;     // A row for this lane (always < M: M % 128 == 0)

  // ---- per-lane loop invariants for the gather A-modes ----
  int g_b = 0, g_oh = 0, g_ow = 0;
  const float* g_src = nullptr; const float* g_dst = nullptr;
  if (AMODE == 1) {
    int hw = am % p.HWin; g_b = am / p.HWin;
    g_oh = hw / p.Win; g_ow = hw - g_oh * p.Win;
  }
  if (AMODE == 3) {
    g_src = p.A + (long)p.e0[am] * 256;
    g_dst = p.A + (long)p.e1[am] * 256;
  }

#if !USE_TDM
  // fallback staging coords: thread t covers col = t/4, 8 bf16 (16B) at k-span [(t%4)*8, +8)
  const int scol = threadIdx.x >> 2;
  const int sks  = (threadIdx.x & 3) * 8;
  const long srow = (long)(nbase + scol) * p.K;
  const bool svalid = (nbase + scol) < p.N;
#endif

  for (int k0 = 0; k0 < p.K; k0 += 32) {
    // ---------- B tile -> LDS ----------
#if USE_TDM
    if (wave == 0) {
      // 2D TDM descriptor: data_size=2B, tile 32(K) x 64(cols), stride K; OOB rows read 0.
      unsigned long long ga = (unsigned long long)(const void*)(p.B + (long)nbase * p.K + k0);
      unsigned lds_off = (unsigned)(unsigned long long)(const void*)&Bs[0][0];
      unsigned dim0 = (unsigned)(p.K - k0);   // remaining K (tile coords relative to tile start)
      unsigned dim1 = (unsigned)(p.N - nbase);
      v4u g0;
      g0[0] = 1u;                                      // count=1 (valid user D#)
      g0[1] = lds_off;                                 // lds_addr
      g0[2] = (unsigned)(ga & 0xffffffffu);            // global_addr lo
      g0[3] = (unsigned)((ga >> 32) & 0x01ffffffu) | (2u << 30);  // addr hi | type=2
      v8i g1;
      g1[0] = (int)(1u << 16);                         // data_size = 2 bytes
      g1[1] = (int)((dim0 & 0xffffu) << 16);           // tensor_dim0 lo
      g1[2] = (int)((dim0 >> 16) | ((dim1 & 0xffffu) << 16)); // dim0 hi | dim1 lo
      g1[3] = (int)((dim1 >> 16) | (32u << 16));       // dim1 hi | tile_dim0 = 32
      g1[4] = 64;                                      // tile_dim1 = 64
      g1[5] = p.K;                                     // tensor_dim0_stride lo
      g1[6] = 0; g1[7] = 0;
      v4i g2 = {0, 0, 0, 0}, g3 = {0, 0, 0, 0};
      v8i g4 = {0, 0, 0, 0, 0, 0, 0, 0};               // extra group (6-arg form): unused
      __builtin_amdgcn_tensor_load_to_lds(g0, g1, g2, g3, g4, 0);
      __builtin_amdgcn_s_wait_tensorcnt(0);
    }
#else
    {
      uint4 q = {0u, 0u, 0u, 0u};
      if (svalid) {
        q = *(const uint4*)(p.B + srow + k0 + sks);
        if (k0 + 32 < p.K) __builtin_prefetch(p.B + srow + k0 + 32 + sks, 0, 0);
      }
      *(uint4*)&Bs[scol][sks] = q;
    }
#endif
    __syncthreads();

    // ---------- A fragment (vectorized per AMODE) ----------
    v16bf a;
    if (AMODE == 0) {
      const float* ap = p.A + (long)am * p.lda + k0;
      const float4* a0 = (const float4*)(ap + ka0);
      const float4* a1 = (const float4*)(ap + ka1);
      cvt8(a, 0, a0[0], a0[1]);
      cvt8(a, 8, a1[0], a1[1]);
    } else if (AMODE == 1) {           // NHWC im2col 3x3 pad1, Cin=128: k = tap*128 + ci
      #pragma unroll
      for (int g = 0; g < 2; ++g) {
        int kg = k0 + (g ? ka1 : ka0);
        int tap = kg >> 7;             // Cin == 128
        int ci  = kg & 127;
        int kh = tap / 3, kw = tap - kh * 3;
        int ih = g_oh + kh - 1, iw = g_ow + kw - 1;
        float4 qa = {0.f,0.f,0.f,0.f}, qb = {0.f,0.f,0.f,0.f};
        if ((unsigned)ih < (unsigned)p.Hin && (unsigned)iw < (unsigned)p.Win) {
          const float4* q = (const float4*)(p.A + ((((long)g_b * p.Hin + ih) * p.Win + iw) << 7) + ci);
          qa = q[0]; qb = q[1];
        }
        cvt8(a, 8 * g, qa, qb);
      }
    } else {                            // edge concat: [h[src] | h[dst]], K = 512
      #pragma unroll
      for (int g = 0; g < 2; ++g) {     // two 8-wide contiguous groups
        int kg = k0 + (g ? ka1 : ka0);
        const float* bp = (kg < 256) ? (g_src + kg) : (g_dst + kg - 256);
        const float4* q = (const float4*)bp;
        cvt8(a, 8 * g, q[0], q[1]);
      }
    }

    // ---------- 4 WMMAs, B fragments from LDS (aligned 128-bit ds loads) ----------
    #pragma unroll
    for (int s = 0; s < 4; ++s) {
      v16bf b = *(const v16bf*)&Bs[s * 16 + l15][kb];
      acc[s] = __builtin_amdgcn_wmma_f32_16x16x32_bf16(
          false, a, false, b, (short)0, acc[s], false, false);
    }
    __syncthreads();
  }

  // ---------- epilogue: C/D layout — element v: row = v + 8*half, col = lane&15 ----------
  #pragma unroll
  for (int s = 0; s < 4; ++s) {
    int col = nbase + s * 16 + l15;
    if (col >= p.N) continue;
    float bv = p.bias ? p.bias[col] : 0.f;
    #pragma unroll
    for (int v = 0; v < 8; ++v) {
      int m = mbase + v + (half ? 8 : 0);
      float val = acc[s][v] + bv;
      if (p.relu) val = fmaxf(val, 0.f);
      if (p.cmode == 0) p.C[(long)m * p.ldc + col] = val;
      else {
        int hw = m % p.HWin; int b = m / p.HWin;
        p.C[((long)b * p.N + col) * p.HWin + hw] = val;
      }
    }
  }
}

// ---------------- GAT attention pieces ----------------
__global__ void attn_proj_kernel(const float* __restrict__ h, const float* __restrict__ as,
                                 const float* __restrict__ ad, float* __restrict__ es,
                                 float* __restrict__ ed, int Nn, int heads, int oc) {
  int tid = blockIdx.x * blockDim.x + threadIdx.x;
  if (tid >= Nn * heads) return;
  int n = tid / heads, hh = tid % heads;
  const float* hv = h + (long)n * heads * oc + (long)hh * oc;
  float s = 0.f, d = 0.f;
  for (int j = 0; j < oc; ++j) { s += hv[j] * as[hh * oc + j]; d += hv[j] * ad[hh * oc + j]; }
  es[tid] = s; ed[tid] = d;
}

__device__ __forceinline__ void atomicMaxF(float* addr, float v) {
  if (v >= 0.f) atomicMax((int*)addr, __float_as_int(v));
  else          atomicMin((unsigned int*)addr, (unsigned int)__float_as_int(v));
}

__device__ __forceinline__ void edge_sd(const int* e0, const int* e1, int e, int& src, int& dst) {
  if (e < NEDG) { src = e0[e]; dst = e1[e]; }
  else          { src = e - NEDG; dst = e - NEDG; }
}

__global__ void edge_scores_kernel(const int* __restrict__ e0, const int* __restrict__ e1,
                                   const float* __restrict__ es, const float* __restrict__ ed,
                                   float* __restrict__ eatt, float* __restrict__ mx, int heads) {
  int tid = blockIdx.x * blockDim.x + threadIdx.x;
  if (tid >= ETOT * heads) return;
  int e = tid / heads, hh = tid % heads;
  int src, dst; edge_sd(e0, e1, e, src, dst);
  float v = es[src * heads + hh] + ed[dst * heads + hh];
  v = (v > 0.f) ? v : 0.2f * v;   // leaky_relu 0.2
  eatt[tid] = v;
  atomicMaxF(&mx[dst * heads + hh], v);
}

__global__ void edge_exp_kernel(const int* __restrict__ e0, const int* __restrict__ e1,
                                float* __restrict__ eatt, const float* __restrict__ mx,
                                float* __restrict__ den, int heads) {
  int tid = blockIdx.x * blockDim.x + threadIdx.x;
  if (tid >= ETOT * heads) return;
  int e = tid / heads, hh = tid % heads;
  int src, dst; edge_sd(e0, e1, e, src, dst);
  float v = __expf(eatt[tid] - mx[dst * heads + hh]);
  eatt[tid] = v;
  atomicAdd(&den[dst * heads + hh], v);
}

__global__ void edge_agg_kernel(const int* __restrict__ e0, const int* __restrict__ e1,
                                const float* __restrict__ eatt, const float* __restrict__ den,
                                const float* __restrict__ h, float* __restrict__ out,
                                int heads, int oc) {
  long tid = (long)blockIdx.x * blockDim.x + threadIdx.x;
  long tot = (long)ETOT * heads * oc;
  if (tid >= tot) return;
  int e = (int)(tid / (heads * oc));
  int r = (int)(tid % (heads * oc));
  int hh = r / oc, j = r % oc;
  int src, dst; edge_sd(e0, e1, e, src, dst);
  float alpha = eatt[(long)e * heads + hh] / (den[dst * heads + hh] + 1e-16f);
  atomicAdd(&out[(long)dst * heads * oc + hh * oc + j],
            h[(long)src * heads * oc + hh * oc + j] * alpha);
}

// ---------------- host-side helpers ----------------
template <int AMODE>
static void launch_gemm(hipStream_t st, const float* A, const __bf16* B, const float* bias,
                        float* C, int M, int N, int K, int lda, int ldc, int cmode,
                        int relu, int Hin, int Win, const int* e0, const int* e1) {
  GemmP p;
  p.A = A; p.B = B; p.bias = bias; p.C = C;
  p.M = M; p.N = N; p.K = K; p.lda = lda; p.ldc = ldc;
  p.cmode = cmode; p.relu = relu;
  p.Hin = Hin; p.Win = Win; p.HWin = Hin * Win;
  p.e0 = e0; p.e1 = e1;
  dim3 grid((N + 63) / 64, (M + 127) / 128);
  gemm_wmma_kernel<AMODE><<<grid, dim3(256), 0, st>>>(p);
}

static void launch_fill(hipStream_t st, float* p, long n, float v) {
  fill_kernel<<<(int)((n + 255) / 256), 256, 0, st>>>(p, n, v);
}

static void launch_cvt(hipStream_t st, const float* in, __bf16* out, long n) {
  f32_to_bf16_kernel<<<(int)((n + 255) / 256), 256, 0, st>>>(in, out, n);
}

extern "C" void kernel_launch(void* const* d_in, const int* in_sizes, int n_in,
                              void* d_out, int out_size, void* d_ws, size_t ws_size,
                              hipStream_t stream) {
  (void)in_sizes; (void)n_in; (void)out_size; (void)ws_size;
  const float* images = (const float*)d_in[0];
  const float* boxes  = (const float*)d_in[1];
  const int*   bidx   = (const int*)d_in[2];
  const int*   ei0    = (const int*)d_in[3];
  const int*   ei1    = ei0 + NEDG;
  const float* wc1 = (const float*)d_in[4];
  const float* g1  = (const float*)d_in[5];  const float* b1 = (const float*)d_in[6];
  const float* wc2 = (const float*)d_in[7];
  const float* g2  = (const float*)d_in[8];  const float* b2 = (const float*)d_in[9];
  const float* wc3 = (const float*)d_in[10];
  const float* g3  = (const float*)d_in[11]; const float* b3 = (const float*)d_in[12];
  const float* wm1 = (const float*)d_in[13]; const float* bm1 = (const float*)d_in[14];
  const float* wm2 = (const float*)d_in[15]; const float* bm2 = (const float*)d_in[16];
  const float* wg1 = (const float*)d_in[17];
  const float* as1 = (const float*)d_in[18]; const float* ad1 = (const float*)d_in[19];
  const float* bi1 = (const float*)d_in[20];
  const float* wg2 = (const float*)d_in[21];
  const float* as2 = (const float*)d_in[22]; const float* ad2 = (const float*)d_in[23];
  const float* bi2 = (const float*)d_in[24];
  const float* wob = (const float*)d_in[25]; const float* bob = (const float*)d_in[26];
  const float* wrl = (const float*)d_in[27]; const float* brl = (const float*)d_in[28];
  float*  ws  = (float*)d_ws;
  __bf16* wbf = (__bf16*)(ws + O_WBF);
  float*  out = (float*)d_out;

  // ---- prep: convert all GEMM weights to bf16 (conv2 also permuted to [N][tap*128+ci]) ----
  permute_wc2_kernel<<<(256 * 1152 + 255) / 256, 256, 0, stream>>>(wc2, wbf + WB_C2);
  launch_cvt(stream, wc3, wbf + WB_C3, 16384);
  launch_cvt(stream, wm1, wbf + WB_M1, 802816);
  launch_cvt(stream, wm2, wbf + WB_M2, 65536);
  launch_cvt(stream, wg1, wbf + WB_G1, 262144);
  launch_cvt(stream, wg2, wbf + WB_G2, 262144);
  launch_cvt(stream, wob, wbf + WB_OB, 38656);
  launch_cvt(stream, wrl, wbf + WB_RL, 26112);

  // ---- conv1 + GN(fs=8) + relu, then Gaussian pool -> NHWC ----
  conv1_kernel<<<(int)((25690112L + 255) / 256), 256, 0, stream>>>(
      images, wc1, ws + O_CONV1, NB, 128, 224, 224);
  launch_fill(stream, ws + O_GN, 128, 0.f);
  gn_accum_kernel<<<dim3(64, 16), 256, 0, stream>>>(ws + O_CONV1, ws + O_GN, NB, 128, 224 * 224, 8);
  gn_apply_kernel<<<(int)((25690112L + 255) / 256), 256, 0, stream>>>(
      ws + O_CONV1, ws + O_GN, g1, b1, NB, 128, 224 * 224, 8);
  gauss_pool_nhwc_kernel<<<(int)((6422528L + 255) / 256), 256, 0, stream>>>(
      ws + O_CONV1, ws + O_POOL1, NB, 128, 224, 224);

  // ---- conv2 via NHWC implicit-GEMM WMMA (M=50176,K=1152,N=256) + GN + relu ----
  launch_gemm<1>(stream, ws + O_POOL1, wbf + WB_C2, nullptr, ws + O_CONV1,
                 NB * 112 * 112, 256, 1152, 0, 0, /*cmode*/1, 0, 112, 112, nullptr, nullptr);
  launch_fill(stream, ws + O_GN, 128, 0.f);
  gn_accum_kernel<<<dim3(64, 32), 256, 0, stream>>>(ws + O_CONV1, ws + O_GN, NB, 256, 112 * 112, 8);
  gn_apply_kernel<<<(int)((12845056L + 255) / 256), 256, 0, stream>>>(
      ws + O_CONV1, ws + O_GN, g2, b2, NB, 256, 112 * 112, 8);
  gauss_pool_nhwc_kernel<<<(int)((3211264L + 255) / 256), 256, 0, stream>>>(
      ws + O_CONV1, ws + O_POOL1, NB, 256, 112, 112);

  // ---- conv3 1x1 on NHWC == plain row-major GEMM (M=12544,K=256,N=64) + GN + relu ----
  launch_gemm<0>(stream, ws + O_POOL1, wbf + WB_C3, nullptr, ws + O_CONV3,
                 NB * 56 * 56, 64, 256, 256, 0, /*cmode*/1, 0, 56, 56, nullptr, nullptr);
  launch_fill(stream, ws + O_GN, 128, 0.f);
  gn_accum_kernel<<<dim3(32, 64), 256, 0, stream>>>(ws + O_CONV3, ws + O_GN, NB, 64, 56 * 56, 1);
  gn_apply_kernel<<<(int)((802816L + 255) / 256), 256, 0, stream>>>(
      ws + O_CONV3, ws + O_GN, g3, b3, NB, 64, 56 * 56, 1);

  // ---- ROI align -> [4096, 3136] (MLP1 A matrix layout) ----
  roi_align_kernel<<<(int)(((long)NBOX * 64 * 49 + 255) / 256), 256, 0, stream>>>(
      ws + O_CONV3, boxes, bidx, ws + O_CONV1, 64, 56, 56);

  // ---- MLP: two WMMA GEMMs ----
  launch_gemm<0>(stream, ws + O_CONV1, wbf + WB_M1, bm1, ws + O_H1, NBOX, 256, 3136, 3136, 256,
                 0, 1, 1, 1, nullptr, nullptr);
  launch_gemm<0>(stream, ws + O_H1, wbf + WB_M2, bm2, ws + O_H2, NBOX, 256, 256, 256, 256,
                 0, 1, 1, 1, nullptr, nullptr);

  // ---- GAT layer 1 (heads=4, oc=256, concat) ----
  launch_gemm<0>(stream, ws + O_H2, wbf + WB_G1, nullptr, ws + O_G1H, NBOX, 1024, 256, 256, 1024,
                 0, 0, 1, 1, nullptr, nullptr);
  attn_proj_kernel<<<(NBOX * 4 + 255) / 256, 256, 0, stream>>>(
      ws + O_G1H, as1, ad1, ws + O_ES1, ws + O_ED1, NBOX, 4, 256);
  launch_fill(stream, ws + O_MAX1, NBOX * 4, -INFINITY);
  launch_fill(stream, ws + O_DEN1, NBOX * 4, 0.f);
  launch_fill(stream, ws + O_AGG1, (long)NBOX * 1024, 0.f);
  edge_scores_kernel<<<(ETOT * 4 + 255) / 256, 256, 0, stream>>>(
      ei0, ei1, ws + O_ES1, ws + O_ED1, ws + O_EATT1, ws + O_MAX1, 4);
  edge_exp_kernel<<<(ETOT * 4 + 255) / 256, 256, 0, stream>>>(
      ei0, ei1, ws + O_EATT1, ws + O_MAX1, ws + O_DEN1, 4);
  edge_agg_kernel<<<(int)(((long)ETOT * 1024 + 255) / 256), 256, 0, stream>>>(
      ei0, ei1, ws + O_EATT1, ws + O_DEN1, ws + O_G1H, ws + O_AGG1, 4, 256);
  bias_add_kernel<<<(int)(((long)NBOX * 1024 + 255) / 256), 256, 0, stream>>>(
      ws + O_AGG1, bi1, (long)NBOX * 1024, 1024);

  // ---- GAT layer 2 (heads=1, oc=256, mean==identity) ----
  launch_gemm<0>(stream, ws + O_AGG1, wbf + WB_G2, nullptr, ws + O_G2H, NBOX, 256, 1024, 1024, 256,
                 0, 0, 1, 1, nullptr, nullptr);
  attn_proj_kernel<<<(NBOX + 255) / 256, 256, 0, stream>>>(
      ws + O_G2H, as2, ad2, ws + O_ES2, ws + O_ED2, NBOX, 1, 256);
  launch_fill(stream, ws + O_MAX2, NBOX, -INFINITY);
  launch_fill(stream, ws + O_DEN2, NBOX, 0.f);
  launch_fill(stream, ws + O_HFIN, (long)NBOX * 256, 0.f);
  edge_scores_kernel<<<(ETOT + 255) / 256, 256, 0, stream>>>(
      ei0, ei1, ws + O_ES2, ws + O_ED2, ws + O_EATT2, ws + O_MAX2, 1);
  edge_exp_kernel<<<(ETOT + 255) / 256, 256, 0, stream>>>(
      ei0, ei1, ws + O_EATT2, ws + O_MAX2, ws + O_DEN2, 1);
  edge_agg_kernel<<<(int)(((long)ETOT * 256 + 255) / 256), 256, 0, stream>>>(
      ei0, ei1, ws + O_EATT2, ws + O_DEN2, ws + O_G2H, ws + O_HFIN, 1, 256);
  bias_add_kernel<<<(int)(((long)NBOX * 256 + 255) / 256), 256, 0, stream>>>(
      ws + O_HFIN, bi2, (long)NBOX * 256, 256);

  // ---- heads: obj logits [4096,151], rel logits [65536,51] (concat-gather A) ----
  launch_gemm<0>(stream, ws + O_HFIN, wbf + WB_OB, bob, out, NBOX, 151, 256, 256, 151,
                 0, 0, 1, 1, nullptr, nullptr);
  launch_gemm<3>(stream, ws + O_HFIN, wbf + WB_RL, brl, out + (long)NBOX * 151, NEDG, 51, 512, 0, 51,
                 0, 0, 1, 1, ei0, ei1);
}
